// EdgeConv_31224412242416
// MI455X (gfx1250) — matrix operand order repaired and checked by reading, not compiled, not run on hardware
//
#include <hip/hip_runtime.h>

typedef __attribute__((ext_vector_type(16))) _Float16 v16h;
typedef __attribute__((ext_vector_type(8)))  _Float16 v8h;
typedef __attribute__((ext_vector_type(8)))  float    v8f;

#define B_    32
#define N_    2048
#define K_    20
#define EPS_  1e-5f

// ---------------------------------------------------------------------------
// Kernel 1: brute-force KNN (top-20 smallest squared distances, self excluded)
// ---------------------------------------------------------------------------
__global__ __launch_bounds__(256) void knn_kernel(const float* __restrict__ X,
                                                  int* __restrict__ idx_out) {
    __shared__ float sx[256], sy[256], sz[256];
    const int b = blockIdx.x >> 3;                  // N_/256 = 8 blocks per batch
    const int q = ((blockIdx.x & 7) << 8) + threadIdx.x;
    const float* Xb = X + (size_t)b * N_ * 3;
    const float qx = Xb[q * 3 + 0];
    const float qy = Xb[q * 3 + 1];
    const float qz = Xb[q * 3 + 2];

    float bd[K_]; int bi[K_];
#pragma unroll
    for (int k = 0; k < K_; ++k) { bd[k] = 3.4e38f; bi[k] = 0; }
    float wd = 3.4e38f; int wi = 0;

    for (int c0 = 0; c0 < N_; c0 += 256) {
        const int cl = c0 + threadIdx.x;
        sx[threadIdx.x] = Xb[cl * 3 + 0];
        sy[threadIdx.x] = Xb[cl * 3 + 1];
        sz[threadIdx.x] = Xb[cl * 3 + 2];
        __syncthreads();
        for (int j = 0; j < 256; ++j) {
            const int c = c0 + j;
            const float dx = qx - sx[j];
            const float dy = qy - sy[j];
            const float dz = qz - sz[j];
            const float d = dx * dx + dy * dy + dz * dz;
            if (c != q && d < wd) {
                // replace current worst (unrolled so arrays stay in VGPRs)
#pragma unroll
                for (int k = 0; k < K_; ++k) if (k == wi) { bd[k] = d; bi[k] = c; }
                wd = -1.0f;
#pragma unroll
                for (int k = 0; k < K_; ++k) if (bd[k] > wd) { wd = bd[k]; wi = k; }
            }
        }
        __syncthreads();
    }
    int* o = idx_out + ((size_t)b * N_ + q) * K_;
#pragma unroll
    for (int k = 0; k < K_; ++k) o[k] = bi[k];
}

// ---------------------------------------------------------------------------
// WMMA helpers (contiguous-K fragment layouts, same k order in A and B, wave32)
// ---------------------------------------------------------------------------
__device__ __forceinline__ v8f wmma32(v16h a, v16h b, v8f c) {
    return __builtin_amdgcn_wmma_f32_16x16x32_f16(false, a, false, b,
                                                  (short)0, c, false, false);
}

// B fragment (KxN = 32x16 chunk): lane holds column n = nt*16 + (lane&15);
// halves are contiguous K = kc*32 + (lane>=16 ? 16 : 0) + 0..15.
__device__ __forceinline__ v16h load_b_frag(const _Float16* __restrict__ w, int ldk,
                                            int nt, int kc, int lh, int hs) {
    const _Float16* p = w + (nt * 16 + lh) * ldk + kc * 32 + hs * 16;
    v8h lo = *(const v8h*)(p);
    v8h hi = *(const v8h*)(p + 8);
    v16h r;
#pragma unroll
    for (int t = 0; t < 8; ++t) { r[t] = lo[t]; r[8 + t] = hi[t]; }
    return r;
}

// A fragment (MxK = 16x32 chunk) from row-major h[m][64]:
// lane holds row m = mt*16 + (lane&15);
// halves are contiguous K = kc*32 + hs*16 + 0..15.
__device__ __forceinline__ v16h load_a_frag(const _Float16* __restrict__ h,
                                            int mt, int kc, int lh, int hs) {
    const _Float16* p = h + (mt * 16 + lh) * 64 + kc * 32 + hs * 16;
    v8h lo = *(const v8h*)(p);
    v8h hi = *(const v8h*)(p + 8);
    v16h r;
#pragma unroll
    for (int t = 0; t < 8; ++t) { r[t] = lo[t]; r[8 + t] = hi[t]; }
    return r;
}

// ---------------------------------------------------------------------------
// Kernel 2: fused edge-feature -> 3x (GEMM + BN + ReLU) -> max over K
// One wave per point: M = 32 rows (20 real edges + 12 pad rows = copies of
// edge 0; ReLU>=0 and duplicates keep the max identical).
// ---------------------------------------------------------------------------
__global__ __launch_bounds__(256) void mlp_kernel(
    const float* __restrict__ X, const int* __restrict__ knn,
    const float* __restrict__ w1, const float* __restrict__ b1,
    const float* __restrict__ g1, const float* __restrict__ be1,
    const float* __restrict__ m1, const float* __restrict__ v1,
    const float* __restrict__ w2, const float* __restrict__ b2,
    const float* __restrict__ g2, const float* __restrict__ be2,
    const float* __restrict__ m2, const float* __restrict__ v2,
    const float* __restrict__ w3, const float* __restrict__ b3,
    const float* __restrict__ g3, const float* __restrict__ be3,
    const float* __restrict__ m3, const float* __restrict__ v3,
    float* __restrict__ out) {

    __shared__ __align__(32) _Float16 wh1[64][32];       // k>=6 zero padded
    __shared__ __align__(32) _Float16 wh2[64][64];
    __shared__ __align__(32) _Float16 wh3[64][64];
    __shared__ float s1[64], t1[64], s2[64], t2[64], s3[64], t3[64];
    __shared__ __align__(32) _Float16 hbuf[8][32 * 64];  // per-wave activation tile

    // ---- stage weights (f32 -> f16) + fused BN affine ----
    for (int e = threadIdx.x; e < 64 * 32; e += 256) {
        const int o = e >> 5, k = e & 31;
        wh1[o][k] = (k < 6) ? (_Float16)w1[o * 6 + k] : (_Float16)0.0f;
    }
    for (int e = threadIdx.x; e < 64 * 64; e += 256) {
        const int o = e >> 6, k = e & 63;
        wh2[o][k] = (_Float16)w2[e];
        wh3[o][k] = (_Float16)w3[e];
    }
    if (threadIdx.x < 64) {
        const int o = threadIdx.x;
        float sv;
        sv = g1[o] * rsqrtf(v1[o] + EPS_); s1[o] = sv; t1[o] = (b1[o] - m1[o]) * sv + be1[o];
        sv = g2[o] * rsqrtf(v2[o] + EPS_); s2[o] = sv; t2[o] = (b2[o] - m2[o]) * sv + be2[o];
        sv = g3[o] * rsqrtf(v3[o] + EPS_); s3[o] = sv; t3[o] = (b3[o] - m3[o]) * sv + be3[o];
    }
    __syncthreads();

    const int wave = threadIdx.x >> 5;
    const int lane = threadIdx.x & 31;
    const int lh   = lane & 15;      // column/row-within-tile index
    const int hs   = lane >> 4;      // half-wave select (K split / M split)
    const int p    = blockIdx.x * 8 + wave;
    const int b    = p >> 11;        // / N_
    const int n    = p & (N_ - 1);
    _Float16* hw = &hbuf[wave][0];

    // ---- layer 1: build A from edge features directly in registers ----
    v16h a1[2];
#pragma unroll
    for (int mt = 0; mt < 2; ++mt) {
        const int m = mt * 16 + lh;
        const int e = (m < K_) ? m : 0;                      // pad rows -> edge 0
        const int j = knn[((size_t)b * N_ + n) * K_ + e];
        const float* xi = X + ((size_t)b * N_ + n) * 3;
        const float* xj = X + ((size_t)b * N_ + j) * 3;
        float f[6];
        f[0] = xi[0]; f[1] = xi[1]; f[2] = xi[2];
        f[3] = xj[0] - xi[0]; f[4] = xj[1] - xi[1]; f[5] = xj[2] - xi[2];
        v16h a;
#pragma unroll
        for (int t = 0; t < 16; ++t) a[t] = (_Float16)0.0f;
#pragma unroll
        for (int t = 0; t < 6; ++t)                          // only K=0..5 nonzero
            a[t] = (hs == 0) ? (_Float16)f[t] : (_Float16)0.0f;
        a1[mt] = a;
    }

    v8f acc[2][4];
#pragma unroll
    for (int mt = 0; mt < 2; ++mt)
#pragma unroll
        for (int nt = 0; nt < 4; ++nt)
#pragma unroll
            for (int t = 0; t < 8; ++t) acc[mt][nt][t] = 0.0f;

#pragma unroll
    for (int nt = 0; nt < 4; ++nt) {
        const v16h bf = load_b_frag(&wh1[0][0], 32, nt, 0, lh, hs);
#pragma unroll
        for (int mt = 0; mt < 2; ++mt) acc[mt][nt] = wmma32(a1[mt], bf, acc[mt][nt]);
    }

    // ---- epilogue 1: BN + ReLU -> f16 LDS tile ----
#pragma unroll
    for (int mt = 0; mt < 2; ++mt)
#pragma unroll
        for (int nt = 0; nt < 4; ++nt) {
            const int o = nt * 16 + lh;
            const float sv = s1[o], tv = t1[o];
#pragma unroll
            for (int r = 0; r < 8; ++r) {
                const int m = mt * 16 + r + hs * 8;
                hw[m * 64 + o] = (_Float16)fmaxf(0.0f, acc[mt][nt][r] * sv + tv);
            }
        }
    asm volatile("s_wait_dscnt 0" ::: "memory");

    // ---- layer 2 ----
#pragma unroll
    for (int mt = 0; mt < 2; ++mt)
#pragma unroll
        for (int nt = 0; nt < 4; ++nt)
#pragma unroll
            for (int t = 0; t < 8; ++t) acc[mt][nt][t] = 0.0f;
#pragma unroll
    for (int kc = 0; kc < 2; ++kc) {
        v16h af[2];
#pragma unroll
        for (int mt = 0; mt < 2; ++mt) af[mt] = load_a_frag(hw, mt, kc, lh, hs);
#pragma unroll
        for (int nt = 0; nt < 4; ++nt) {
            const v16h bf = load_b_frag(&wh2[0][0], 64, nt, kc, lh, hs);
#pragma unroll
            for (int mt = 0; mt < 2; ++mt) acc[mt][nt] = wmma32(af[mt], bf, acc[mt][nt]);
        }
    }
    asm volatile("s_wait_dscnt 0" ::: "memory");
#pragma unroll
    for (int mt = 0; mt < 2; ++mt)
#pragma unroll
        for (int nt = 0; nt < 4; ++nt) {
            const int o = nt * 16 + lh;
            const float sv = s2[o], tv = t2[o];
#pragma unroll
            for (int r = 0; r < 8; ++r) {
                const int m = mt * 16 + r + hs * 8;
                hw[m * 64 + o] = (_Float16)fmaxf(0.0f, acc[mt][nt][r] * sv + tv);
            }
        }
    asm volatile("s_wait_dscnt 0" ::: "memory");

    // ---- layer 3 ----
#pragma unroll
    for (int mt = 0; mt < 2; ++mt)
#pragma unroll
        for (int nt = 0; nt < 4; ++nt)
#pragma unroll
            for (int t = 0; t < 8; ++t) acc[mt][nt][t] = 0.0f;
#pragma unroll
    for (int kc = 0; kc < 2; ++kc) {
        v16h af[2];
#pragma unroll
        for (int mt = 0; mt < 2; ++mt) af[mt] = load_a_frag(hw, mt, kc, lh, hs);
#pragma unroll
        for (int nt = 0; nt < 4; ++nt) {
            const v16h bf = load_b_frag(&wh3[0][0], 64, nt, kc, lh, hs);
#pragma unroll
            for (int mt = 0; mt < 2; ++mt) acc[mt][nt] = wmma32(af[mt], bf, acc[mt][nt]);
        }
    }

    // ---- epilogue 3: BN + ReLU + max over the 32 rows -> out ----
    float* outp = out + ((size_t)b * N_ + n) * 64;
#pragma unroll
    for (int nt = 0; nt < 4; ++nt) {
        const int o = nt * 16 + lh;
        const float sv = s3[o], tv = t3[o];
        float mx = 0.0f;                     // ReLU output >= 0
#pragma unroll
        for (int mt = 0; mt < 2; ++mt)
#pragma unroll
            for (int r = 0; r < 8; ++r)
                mx = fmaxf(mx, fmaxf(0.0f, acc[mt][nt][r] * sv + tv));
        // combine the two half-wave row groups (same o, other 16 rows)
        mx = fmaxf(mx, __shfl_xor(mx, 16, 32));
        if (hs == 0) outp[o] = mx;
    }
}

// ---------------------------------------------------------------------------
extern "C" void kernel_launch(void* const* d_in, const int* in_sizes, int n_in,
                              void* d_out, int out_size, void* d_ws, size_t ws_size,
                              hipStream_t stream) {
    (void)in_sizes; (void)n_in; (void)out_size; (void)ws_size;
    const float* X   = (const float*)d_in[0];
    const float* w1  = (const float*)d_in[1];
    const float* b1  = (const float*)d_in[2];
    const float* g1  = (const float*)d_in[3];
    const float* be1 = (const float*)d_in[4];
    const float* m1  = (const float*)d_in[5];
    const float* v1  = (const float*)d_in[6];
    const float* w2  = (const float*)d_in[7];
    const float* b2  = (const float*)d_in[8];
    const float* g2  = (const float*)d_in[9];
    const float* be2 = (const float*)d_in[10];
    const float* m2  = (const float*)d_in[11];
    const float* v2  = (const float*)d_in[12];
    const float* w3  = (const float*)d_in[13];
    const float* b3  = (const float*)d_in[14];
    const float* g3  = (const float*)d_in[15];
    const float* be3 = (const float*)d_in[16];
    const float* m3  = (const float*)d_in[17];
    const float* v3  = (const float*)d_in[18];

    int* knn_idx = (int*)d_ws;               // B*N*K ints = 5.24 MB

    knn_kernel<<<B_ * (N_ / 256), 256, 0, stream>>>(X, knn_idx);
    mlp_kernel<<<(B_ * N_) / 8, 256, 0, stream>>>(
        X, knn_idx,
        w1, b1, g1, be1, m1, v1,
        w2, b2, g2, be2, m2, v2,
        w3, b3, g3, be3, m3, v3,
        (float*)d_out);
}
